// SeparableAttention_9320079032978
// MI455X (gfx1250) — matrix operand order repaired and checked by reading, not compiled, not run on hardware
//
#include <hip/hip_runtime.h>

#define Bn 4
#define Cn 192
#define Sn 64
#define Ln 512
#define CH_STRIDE (Sn * Ln)          // 32768
#define BATCH_STRIDE (Cn * Sn * Ln)  // 6291456

typedef __attribute__((ext_vector_type(16))) __bf16 v16bf;
typedef __attribute__((ext_vector_type(8)))  float  v8f;

union ABTile { v16bf v; unsigned int d[8]; uint4 q[2]; };

__device__ inline unsigned short f2bf(float f) {
    union { float f; unsigned int u; } x;
    x.f = f;
    unsigned int r = x.u + 0x7FFFu + ((x.u >> 16) & 1u);  // round-to-nearest-even
    return (unsigned short)(r >> 16);
}
__device__ inline unsigned int pk2bf(float lo, float hi) {
    return (unsigned int)f2bf(lo) | ((unsigned int)f2bf(hi) << 16);
}

__device__ inline float waveMax(float v) {
    #pragma unroll
    for (int o = 16; o > 0; o >>= 1) v = fmaxf(v, __shfl_down(v, o, 32));
    return v;
}
__device__ inline float waveSum(float v) {
    #pragma unroll
    for (int o = 16; o > 0; o >>= 1) v += __shfl_down(v, o, 32);
    return v;
}

// B-operand LDS layout (dword units): dword vv of column col holds bf16 rows
// (32*kk + 16*khalf + 2*vv, +1). A lane's full fragment = 8 contiguous dwords.
__device__ inline int bIdx(int kk, int khalf, int col, int vv) {
    return ((kk * 2 + khalf) * 64 + col) * 8 + vv;
}

// ---------------------------------------------------------------------------
// Kernel W: pre-swizzle Wv (in_proj rows 193..384) and Wo into bf16 A-operand
// layout for v_wmma_f32_16x16x32_bf16:
//   tile t (16 rows) x kstep kk (32 K): dword index ((t*6+kk)*32+lane)*8+v
//   lane: m=lane&15 (row), kh=lane>>4 ; dword v: K = 2*v+16*kh
// ---------------------------------------------------------------------------
__global__ void pack_weights_kernel(const float* __restrict__ in_w,
                                    const float* __restrict__ out_w,
                                    unsigned int* __restrict__ wvp,
                                    unsigned int* __restrict__ wop) {
    int gid = blockIdx.x * 256 + threadIdx.x;
    const int per_mat = 12 * 6 * 32 * 8;  // 18432 dwords
    if (gid >= 2 * per_mat) return;
    int mat = gid / per_mat;
    int r   = gid % per_mat;
    int v    = r & 7;
    int lane = (r >> 3) & 31;
    int rest = r >> 8;
    int kk   = rest % 6;
    int t    = rest / 6;
    int m  = lane & 15;
    int kh = lane >> 4;
    int o  = t * 16 + m;
    int K  = 2 * v + 16 * kh + 32 * kk;
    const float* src = (mat == 0) ? (in_w + (1 + Cn + o) * Cn) : (out_w + o * Cn);
    unsigned int d = pk2bf(src[K], src[K + 1]);
    if (mat == 0) wvp[r] = d; else wop[r] = d;
}

// ---------------------------------------------------------------------------
// Kernel A: per (b,s): q projection + softmax over L + score-weighted key
// reduction + Wk projection  ->  context[b][s][0..191]
// ---------------------------------------------------------------------------
__global__ __launch_bounds__(512) void scores_context_kernel(
    const float* __restrict__ query, const float* __restrict__ key,
    const float* __restrict__ in_w, const float* __restrict__ in_b,
    float* __restrict__ ctx_out) {
    const int s = blockIdx.x, b = blockIdx.y;
    const int tid  = threadIdx.x;              // 0..511, thread = l
    const int lane = tid & 31, wv = tid >> 5;  // 16 waves

    __shared__ float wq[Cn];
    __shared__ float scores[Ln];
    __shared__ float kbar[Cn];
    __shared__ float red[16];
    __shared__ float bcast;

    if (tid < Cn) wq[tid] = in_w[tid];  // row 0 of (385,192)
    __syncthreads();

    const int base = b * BATCH_STRIDE + s * Ln;

    // q projection (coalesced across l)
    float acc = in_b[0];
    const float* qp = query + base + tid;
    #pragma unroll 4
    for (int c = 0; c < Cn; ++c) acc += wq[c] * qp[c * CH_STRIDE];

    // block softmax over 512
    float m = waveMax(acc);
    if (lane == 0) red[wv] = m;
    __syncthreads();
    if (tid == 0) {
        float mm = red[0];
        for (int i = 1; i < 16; ++i) mm = fmaxf(mm, red[i]);
        bcast = mm;
    }
    __syncthreads();
    float e = __expf(acc - bcast);
    float ps = waveSum(e);
    if (lane == 0) red[wv] = ps;
    __syncthreads();
    if (tid == 0) {
        float t = 0.f;
        for (int i = 0; i < 16; ++i) t += red[i];
        bcast = t;
    }
    __syncthreads();
    scores[tid] = e / bcast;
    __syncthreads();

    // kbar[c] = sum_l key[b,c,s,l] * scores[l]  (wave wv owns 12 channels)
    #pragma unroll
    for (int i = 0; i < 12; ++i) {
        int c = wv * 12 + i;
        const float* kp = key + base + c * CH_STRIDE;
        float sum = 0.f;
        #pragma unroll
        for (int t = 0; t < 16; ++t) {
            int l = lane + t * 32;
            sum += kp[l] * scores[l];
        }
        sum = waveSum(sum);
        if (lane == 0) kbar[c] = sum;
    }
    __syncthreads();

    // context[o] = bk[o] + Wk[o,:] . kbar   (softmax weights sum to 1)
    if (tid < Cn) {
        float cx = in_b[1 + tid];
        const float* wr = in_w + (1 + tid) * Cn;
        #pragma unroll 4
        for (int c = 0; c < Cn; ++c) cx += wr[c] * kbar[c];
        ctx_out[(b * Sn + s) * Cn + tid] = cx;
    }
}

// ---------------------------------------------------------------------------
// Kernel B: out = Wo @ (relu(Wv @ value + bv) * context) + bo  via bf16 WMMA
// block = 384 threads = 12 waves = 12 o-tiles; per (b, s, 64-wide l tile)
// ---------------------------------------------------------------------------
__global__ __launch_bounds__(384) void attn_out_kernel(
    const float* __restrict__ value, const float* __restrict__ in_b,
    const float* __restrict__ out_b, const float* __restrict__ ctx_in,
    const unsigned int* __restrict__ wvp, const unsigned int* __restrict__ wop,
    float* __restrict__ out) {
    const int l0 = blockIdx.x * 64, s = blockIdx.y, b = blockIdx.z;
    const int tid = threadIdx.x;

    __shared__ unsigned int Vt[6 * 2 * 64 * 8];  // value tile, B-operand layout (24 KB)
    __shared__ unsigned int Xt[6 * 2 * 64 * 8];  // intermediate tile, same layout (24 KB)
    __shared__ float ctx[Cn], bv[Cn], bo[Cn];

    const int base = b * BATCH_STRIDE + s * Ln + l0;

    // cooperative load of value tile: channel pairs -> packed bf16 dword in
    // B-operand layout (one ds_store_b32 per dword, global loads coalesced)
    #pragma unroll 4
    for (int idx = tid; idx < 96 * 64; idx += 384) {
        int p = idx >> 6, l = idx & 63;
        int c = 2 * p;
        float x0 = value[base + c * CH_STRIDE + l];
        float x1 = value[base + (c + 1) * CH_STRIDE + l];
        int kk = c >> 5, w32 = c & 31;
        int khh = w32 >> 4, vv = (w32 & 15) >> 1;
        Vt[bIdx(kk, khh, l, vv)] = pk2bf(x0, x1);
    }
    if (tid < Cn) {
        ctx[tid] = ctx_in[(b * Sn + s) * Cn + tid];
        bv[tid]  = in_b[1 + Cn + tid];
        bo[tid]  = out_b[tid];
    }
    __syncthreads();

    const int w = tid >> 5;  // wave id = o-tile (0..11)
    const int lane = tid & 31;
    const int m = lane & 15, kh = lane >> 4;
    const int rb = w * 16 + 8 * kh;  // this lane's 8 accumulator rows: rb..rb+7

    // X-store mapping: rows rb..rb+7 -> dwords vv0..vv0+3 of (kk2, kh2)
    const int kk2 = rb >> 5, rb32 = rb & 31;
    const int kh2 = rb32 >> 4, vv0 = (rb32 & 15) >> 1;  // 0 or 4

    // hoist per-row constants into registers (j-invariant): 2x ds_load_b128 each
    float ctxr[8], bvr[8], bor[8];
    {
        float4 c0 = *(const float4*)&ctx[rb], c1 = *(const float4*)&ctx[rb + 4];
        float4 v0 = *(const float4*)&bv[rb],  v1 = *(const float4*)&bv[rb + 4];
        float4 o0 = *(const float4*)&bo[rb],  o1 = *(const float4*)&bo[rb + 4];
        ctxr[0] = c0.x; ctxr[1] = c0.y; ctxr[2] = c0.z; ctxr[3] = c0.w;
        ctxr[4] = c1.x; ctxr[5] = c1.y; ctxr[6] = c1.z; ctxr[7] = c1.w;
        bvr[0]  = v0.x; bvr[1]  = v0.y; bvr[2]  = v0.z; bvr[3]  = v0.w;
        bvr[4]  = v1.x; bvr[5]  = v1.y; bvr[6]  = v1.z; bvr[7]  = v1.w;
        bor[0]  = o0.x; bor[1]  = o0.y; bor[2]  = o0.z; bor[3]  = o0.w;
        bor[4]  = o1.x; bor[5]  = o1.y; bor[6]  = o1.z; bor[7]  = o1.w;
    }

    ABTile A[6];

    // ---- GEMM1: X = relu(Wv @ V + bv) * ctx ----
    {
        const uint4* p = (const uint4*)wvp;
        #pragma unroll
        for (int kk = 0; kk < 6; ++kk) {
            int bi = ((w * 6 + kk) * 32 + lane) * 2;
            A[kk].q[0] = p[bi];
            A[kk].q[1] = p[bi + 1];
        }
    }

    #pragma unroll
    for (int j = 0; j < 4; ++j) {
        int n0 = j * 16;
        v8f acc;
        #pragma unroll
        for (int r = 0; r < 8; ++r) acc[r] = bvr[r];
        #pragma unroll
        for (int kk = 0; kk < 6; ++kk) {
            ABTile Bm;
            const uint4* q = (const uint4*)&Vt[bIdx(kk, kh, n0 + m, 0)];
            Bm.q[0] = q[0];
            Bm.q[1] = q[1];
            acc = __builtin_amdgcn_wmma_f32_16x16x32_bf16(
                false, A[kk].v, false, Bm.v, (short)0, acc, false, false);
        }
        // branchless epilogue: relu * context, pack to bf16, one ds_store_b128
        uint4 px;
        {
            float x[8];
            #pragma unroll
            for (int r = 0; r < 8; ++r) x[r] = fmaxf(acc[r], 0.f) * ctxr[r];
            px.x = pk2bf(x[0], x[1]);
            px.y = pk2bf(x[2], x[3]);
            px.z = pk2bf(x[4], x[5]);
            px.w = pk2bf(x[6], x[7]);
        }
        *(uint4*)&Xt[bIdx(kk2, kh2, n0 + m, vv0)] = px;
    }
    __syncthreads();

    // ---- GEMM2: out = Wo @ X + bo ----
    {
        const uint4* p = (const uint4*)wop;
        #pragma unroll
        for (int kk = 0; kk < 6; ++kk) {
            int bi = ((w * 6 + kk) * 32 + lane) * 2;
            A[kk].q[0] = p[bi];
            A[kk].q[1] = p[bi + 1];
        }
    }

    #pragma unroll
    for (int j = 0; j < 4; ++j) {
        int n0 = j * 16;
        v8f acc;
        #pragma unroll
        for (int r = 0; r < 8; ++r) acc[r] = bor[r];
        #pragma unroll
        for (int kk = 0; kk < 6; ++kk) {
            ABTile Bm;
            const uint4* q = (const uint4*)&Xt[bIdx(kk, kh, n0 + m, 0)];
            Bm.q[0] = q[0];
            Bm.q[1] = q[1];
            acc = __builtin_amdgcn_wmma_f32_16x16x32_bf16(
                false, A[kk].v, false, Bm.v, (short)0, acc, false, false);
        }
        #pragma unroll
        for (int r = 0; r < 8; ++r) {
            out[base + (rb + r) * CH_STRIDE + n0 + m] = acc[r];
        }
    }
}

// ---------------------------------------------------------------------------
extern "C" void kernel_launch(void* const* d_in, const int* in_sizes, int n_in,
                              void* d_out, int out_size, void* d_ws, size_t ws_size,
                              hipStream_t stream) {
    const float* query = (const float*)d_in[0];
    const float* key   = (const float*)d_in[1];
    const float* value = (const float*)d_in[2];
    const float* in_w  = (const float*)d_in[3];   // (385,192)
    const float* in_b  = (const float*)d_in[4];   // (385,)
    const float* out_w = (const float*)d_in[5];   // (192,192)
    const float* out_b = (const float*)d_in[6];   // (192,)
    float* out = (float*)d_out;

    // workspace layout
    float* ctx_ws = (float*)d_ws;  // B*S*C f32 = 192 KB
    char*  wbase  = (char*)d_ws + (size_t)Bn * Sn * Cn * sizeof(float);
    unsigned int* wvp = (unsigned int*)wbase;  // 73728 B
    unsigned int* wop = (unsigned int*)(wbase + 12 * 6 * 32 * 8 * 4);

    pack_weights_kernel<<<(2 * 12 * 6 * 32 * 8 + 255) / 256, 256, 0, stream>>>(
        in_w, out_w, wvp, wop);

    scores_context_kernel<<<dim3(Sn, Bn, 1), 512, 0, stream>>>(
        query, key, in_w, in_b, ctx_ws);

    attn_out_kernel<<<dim3(Ln / 64, Sn, Bn), 384, 0, stream>>>(
        value, in_b, out_b, ctx_ws, wvp, wop, out);
}
